// SlideGraphGNN_4157528342782
// MI455X (gfx1250) — matrix operand order repaired and checked by reading, not compiled, not run on hardware
//
#include <hip/hip_runtime.h>
#include <hip/hip_bf16.h>
#include <math.h>

typedef __attribute__((ext_vector_type(16))) _Float16 v16h;
typedef __attribute__((ext_vector_type(8)))  float    v8f;

#define BN_EPS 1e-5f

// ---------------------------------------------------------------- helpers
__device__ __forceinline__ float gelu_exact(float v) {
    // approximate=False GELU: 0.5*x*(1+erf(x/sqrt(2)))
    return 0.5f * v * (1.0f + erff(v * 0.70710678118654752440f));
}

// order-preserving float<->uint encoding for atomic max
__device__ __forceinline__ unsigned fenc(float f) {
    unsigned u = __float_as_uint(f);
    return (u & 0x80000000u) ? ~u : (u | 0x80000000u);
}
__device__ __forceinline__ float fdec(unsigned k) {
    return (k & 0x80000000u) ? __uint_as_float(k & 0x7FFFFFFFu)
                             : __uint_as_float(~k);
}

// ---------------------------------------------------------------- W0 -> f16, pre-swizzled into per-lane WMMA B fragments
// B fragment k order chosen by this code (32x16, KxN), same as the A fragment:
//   lane l: col = l&15, half = l>>4 ; element j (0..15): K = kb*32 + half*16 + j
// Flat index: Wswz[kb*512 + lane*16 + j]
__global__ void convert_w0_kernel(const float* __restrict__ W0, _Float16* __restrict__ Wswz,
                                  int F) {
    int idx = blockIdx.x * 256 + threadIdx.x;          // F*16 elements
    if (idx >= F * 16) return;
    int kb   = idx >> 9;           // /512 (one kb = 32x16 tile = 512 halfs)
    int r    = idx & 511;
    int lane = r >> 4;
    int j    = r & 15;
    int hs   = lane >> 4;
    int col  = lane & 15;
    int K    = kb * 32 + hs * 16 + j;
    Wswz[idx] = (_Float16)W0[K * 16 + col];
}

// ---------------------------------------------------------------- GEMM0: pre0 = x @ W0 + b0  (WMMA f16, f32 accum)
// block = 256 threads = 8 waves; each wave computes a 16-row x 16-col tile, K=F in steps of 32.
// Also emits deterministic per-block BN partial sums (sum, sumsq per feature).
__global__ void __launch_bounds__(256)
gemm0_wmma_kernel(const float* __restrict__ x, const _Float16* __restrict__ Wswz,
                  const float* __restrict__ b0, float* __restrict__ pre0,
                  float* __restrict__ partials, int Nn, int F) {
    __shared__ _Float16 Wlds[512 * 16];   // up to F=512 -> 16KB of B fragments
    __shared__ float red[512];            // 256 sums + 256 sumsq

    const int t    = threadIdx.x;
    const int nfr  = F * 16;              // total halfs
    // cooperative 16B-chunk copy of swizzled W into LDS
    const uint4* wg = reinterpret_cast<const uint4*>(Wswz);
    uint4*       wl = reinterpret_cast<uint4*>(Wlds);
    for (int i = t; i < nfr / 8; i += 256) wl[i] = wg[i];
    __syncthreads();

    const int wave    = t >> 5;
    const int lane    = t & 31;
    const int rowBase = (blockIdx.x * 8 + wave) * 16;
    const int row     = lane & 15;
    const int hs      = lane >> 4;
    const int col     = lane & 15;
    const bool active = (rowBase + 16) <= Nn;          // N % 16 == 0 for this problem

    float s = 0.0f, ss = 0.0f;
    if (active) {
        const float* xrow = x + (size_t)(rowBase + row) * (size_t)F;
        v8f c = {};
        const int ksteps = F >> 5;
        for (int kb = 0; kb < ksteps; ++kb) {
            // A fragment: lane holds row `row`, K = kb*32 + hs*16 + {0..15}
            const int kA = kb * 32 + hs * 16;
            float4 p0 = *reinterpret_cast<const float4*>(xrow + kA);
            float4 p1 = *reinterpret_cast<const float4*>(xrow + kA + 4);
            float4 q0 = *reinterpret_cast<const float4*>(xrow + kA + 8);
            float4 q1 = *reinterpret_cast<const float4*>(xrow + kA + 12);
            v16h a;
            a[0]=(_Float16)p0.x; a[1]=(_Float16)p0.y; a[2]=(_Float16)p0.z; a[3]=(_Float16)p0.w;
            a[4]=(_Float16)p1.x; a[5]=(_Float16)p1.y; a[6]=(_Float16)p1.z; a[7]=(_Float16)p1.w;
            a[8]=(_Float16)q0.x; a[9]=(_Float16)q0.y; a[10]=(_Float16)q0.z; a[11]=(_Float16)q0.w;
            a[12]=(_Float16)q1.x; a[13]=(_Float16)q1.y; a[14]=(_Float16)q1.z; a[15]=(_Float16)q1.w;
            // B fragment: contiguous 32B per lane (pre-swizzled)
            v16h b = *reinterpret_cast<const v16h*>(&Wlds[kb * 512 + lane * 16]);
            c = __builtin_amdgcn_wmma_f32_16x16x32_f16(
                    false, a, false, b, (short)0, c, false, false);
        }
        // C/D layout: lane: col = lane&15, rows = hs*8 + r (r = vgpr index)
        const float bias = b0[col];
        #pragma unroll
        for (int r = 0; r < 8; ++r) {
            float v = c[r] + bias;
            pre0[(size_t)(rowBase + hs * 8 + r) * 16 + col] = v;
            s += v; ss += v * v;
        }
    }
    // deterministic per-block BN partials: feature f = t&15
    red[t]       = s;
    red[256 + t] = ss;
    __syncthreads();
    if (t < 16) {
        float acc = 0.0f, acc2 = 0.0f;
        for (int i = t; i < 256; i += 16) { acc += red[i]; acc2 += red[256 + i]; }
        partials[blockIdx.x * 32 + t]      = acc;
        partials[blockIdx.x * 32 + 16 + t] = acc2;
    }
}

// ---------------------------------------------------------------- BN finalize: alpha/beta per feature (deterministic)
__global__ void bn_finalize_kernel(const float* __restrict__ partials, int nblocks, int nF,
                                   float invN, const float* __restrict__ g,
                                   const float* __restrict__ be, float* __restrict__ ab) {
    int f = threadIdx.x;
    if (f >= nF) return;
    float s = 0.0f, ss = 0.0f;
    for (int b = 0; b < nblocks; ++b) {
        s  += partials[b * 2 * nF + f];
        ss += partials[b * 2 * nF + nF + f];
    }
    float mean = s * invN;
    float var  = ss * invN - mean * mean;
    var = var < 0.0f ? 0.0f : var;
    float alpha = g[f] * rsqrtf(var + BN_EPS);
    ab[f]      = alpha;
    ab[nF + f] = be[f] - mean * alpha;
}

// ---------------------------------------------------------------- layer post: BN apply (+GELU), store h, project to 8, Z_sum, pool-max
template <int NH, bool FIRST>
__global__ void __launch_bounds__(256)
layer_post_kernel(const float* __restrict__ pre, const float* __restrict__ ab,
                  const float* __restrict__ Wl, const float* __restrict__ bl,
                  const int* __restrict__ batch, float* __restrict__ hout,
                  float* __restrict__ zsum, unsigned* __restrict__ pool, int Nn) {
    __shared__ float al[NH], bt[NH], Ws[NH * 8], bs[8];
    __shared__ unsigned pmax[16 * 8];
    const int t = threadIdx.x;
    if (t < NH) { al[t] = ab[t]; bt[t] = ab[NH + t]; }
    for (int i = t; i < NH * 8; i += 256) Ws[i] = Wl[i];
    if (t < 8) bs[t] = bl[t];
    if (t < 128) pmax[t] = 0u;
    __syncthreads();

    const int n = blockIdx.x * 256 + t;
    if (n < Nn) {
        float hv[NH];
        #pragma unroll
        for (int f = 0; f < NH; ++f) {
            float v = al[f] * pre[(size_t)n * NH + f] + bt[f];
            if (FIRST) v = gelu_exact(v);
            hv[f] = v;
            hout[(size_t)n * NH + f] = v;
        }
        const int g = batch[n];
        #pragma unroll
        for (int j = 0; j < 8; ++j) {
            float z = bs[j];
            #pragma unroll
            for (int f = 0; f < NH; ++f) z += hv[f] * Ws[f * 8 + j];
            if (FIRST) z = gelu_exact(z);
            if (FIRST) zsum[(size_t)n * 8 + j]  = z;
            else       zsum[(size_t)n * 8 + j] += z;
            atomicMax(&pmax[g * 8 + j], fenc(z));
        }
    }
    __syncthreads();
    if (t < 128) atomicMax(&pool[t], pmax[t]);
}

// ---------------------------------------------------------------- edge scatter: agg[dst] += h[src]   (h is 16 wide)
__global__ void __launch_bounds__(256)
edge_scatter_kernel(const int* __restrict__ ei, const float* __restrict__ h,
                    float* __restrict__ agg, int En) {
    const int t = blockIdx.x * 256 + threadIdx.x;
    const int e = t >> 2, c = t & 3;
    if (e >= En) return;
    const int s = ei[e];
    const int d = ei[En + e];
    const float4 v = *reinterpret_cast<const float4*>(h + (size_t)s * 16 + c * 4);
    float* ap = agg + (size_t)d * 16 + c * 4;
    atomicAdd(ap + 0, v.x);
    atomicAdd(ap + 1, v.y);
    atomicAdd(ap + 2, v.z);
    atomicAdd(ap + 3, v.w);
}

// ---------------------------------------------------------------- GIN pre: pre = (h + agg) @ W + b, + BN partials
template <int NOUT>
__global__ void __launch_bounds__(256)
gin_pre_kernel(const float* __restrict__ h, const float* __restrict__ agg,
               const float* __restrict__ W, const float* __restrict__ bias,
               float* __restrict__ pre, float* __restrict__ partials, int Nn) {
    __shared__ float Ws[16 * NOUT];
    __shared__ float bs[NOUT];
    __shared__ float red[256 * NOUT];
    __shared__ float red2[256 * NOUT];
    const int t = threadIdx.x;
    for (int i = t; i < 16 * NOUT; i += 256) Ws[i] = W[i];
    if (t < NOUT) bs[t] = bias[t];
    __syncthreads();

    const int n = blockIdx.x * 256 + t;
    float out[NOUT];
    if (n < Nn) {
        float hv[16];
        #pragma unroll
        for (int f = 0; f < 16; ++f)
            hv[f] = h[(size_t)n * 16 + f] + agg[(size_t)n * 16 + f];
        #pragma unroll
        for (int o = 0; o < NOUT; ++o) {
            float a = bs[o];
            #pragma unroll
            for (int f = 0; f < 16; ++f) a += hv[f] * Ws[f * NOUT + o];
            out[o] = a;
            pre[(size_t)n * NOUT + o] = a;
        }
    } else {
        #pragma unroll
        for (int o = 0; o < NOUT; ++o) out[o] = 0.0f;
    }
    #pragma unroll
    for (int o = 0; o < NOUT; ++o) {
        red [o * 256 + t] = out[o];
        red2[o * 256 + t] = out[o] * out[o];
    }
    __syncthreads();
    for (int s = 128; s > 0; s >>= 1) {
        if (t < s) {
            #pragma unroll
            for (int o = 0; o < NOUT; ++o) {
                red [o * 256 + t] += red [o * 256 + t + s];
                red2[o * 256 + t] += red2[o * 256 + t + s];
            }
        }
        __syncthreads();
    }
    if (t < NOUT) {
        partials[blockIdx.x * 2 * NOUT + t]        = red [t * 256];
        partials[blockIdx.x * 2 * NOUT + NOUT + t] = red2[t * 256];
    }
}

// ---------------------------------------------------------------- combine 3 pooled-max tables -> out[16,8]
__global__ void pool_combine_kernel(const unsigned* __restrict__ p, float* __restrict__ out) {
    const int t = threadIdx.x;
    if (t < 128) out[t] = fdec(p[t]) + fdec(p[128 + t]) + fdec(p[256 + t]);
}

// ---------------------------------------------------------------- launch
extern "C" void kernel_launch(void* const* d_in, const int* in_sizes, int n_in,
                              void* d_out, int out_size, void* d_ws, size_t ws_size,
                              hipStream_t stream) {
    // inputs (setup_inputs order)
    const float* x    = (const float*)d_in[0];
    const int*   ei   = (const int*)  d_in[1];
    const int*   bat  = (const int*)  d_in[2];
    const float* W0   = (const float*)d_in[3];
    const float* b0   = (const float*)d_in[4];
    const float* g0   = (const float*)d_in[5];
    const float* be0  = (const float*)d_in[6];
    const float* Wl0  = (const float*)d_in[7];
    const float* bl0  = (const float*)d_in[8];
    const float* W1   = (const float*)d_in[9];
    const float* b1   = (const float*)d_in[10];
    const float* g1   = (const float*)d_in[11];
    const float* be1  = (const float*)d_in[12];
    const float* Wl1  = (const float*)d_in[13];
    const float* bl1  = (const float*)d_in[14];
    const float* W2   = (const float*)d_in[15];
    const float* b2   = (const float*)d_in[16];
    const float* g2   = (const float*)d_in[17];
    const float* be2  = (const float*)d_in[18];
    const float* Wl2  = (const float*)d_in[19];
    const float* bl2  = (const float*)d_in[20];

    const int N = in_sizes[2];          // 100000
    const int E = in_sizes[1] / 2;      // 3200000
    const int F = in_sizes[0] / N;      // 512

    float* out_pool = (float*)d_out;                 // [16*8]
    float* out_zsum = (float*)d_out + 128;           // [N*8]
    float* out_h    = (float*)d_out + 128 + (size_t)N * 8;   // [N*8]

    // workspace carve-out (256B aligned)
    char*  ws  = (char*)d_ws;
    size_t off = 0;
    auto carve = [&](size_t bytes) -> void* {
        off = (off + 255) & ~(size_t)255;
        void* p = ws + off;
        off += bytes;
        return p;
    };
    _Float16* Wswz  = (_Float16*)carve((size_t)F * 16 * 2);
    float*    pre   = (float*)   carve((size_t)N * 16 * 4);
    float*    h0    = (float*)   carve((size_t)N * 16 * 4);
    float*    h1    = (float*)   carve((size_t)N * 16 * 4);
    float*    agg   = (float*)   carve((size_t)N * 16 * 4);
    const int WAVES = (N + 15) / 16;
    const int GB    = (WAVES + 7) / 8;               // gemm0 blocks
    const int NB    = (N + 255) / 256;               // per-node blocks
    float*    partA = (float*)   carve((size_t)GB * 32 * 4);
    float*    partE = (float*)   carve((size_t)NB * 32 * 4);
    float*    ab0   = (float*)   carve(32 * 4);
    float*    ab1   = (float*)   carve(32 * 4);
    float*    ab2   = (float*)   carve(16 * 4);
    unsigned* pool3 = (unsigned*)carve(3 * 128 * 4);

    const float invN = 1.0f / (float)N;

    // prep: swizzle W0 -> f16 fragments; reset pooled-max accumulators
    hipMemsetAsync(pool3, 0, 3 * 128 * 4, stream);
    convert_w0_kernel<<<(F * 16 + 255) / 256, 256, 0, stream>>>(W0, Wswz, F);

    // layer 0: pre0 = x@W0 + b0 (WMMA), BN stats, BN+GELU, project, pool
    gemm0_wmma_kernel<<<GB, 256, 0, stream>>>(x, Wswz, b0, pre, partA, N, F);
    bn_finalize_kernel<<<1, 32, 0, stream>>>(partA, GB, 16, invN, g0, be0, ab0);
    layer_post_kernel<16, true><<<NB, 256, 0, stream>>>(
        pre, ab0, Wl0, bl0, bat, h0, out_zsum, pool3, N);

    // layer 1: agg = segsum(h0[src]->dst); pre1 = (h0+agg)@W1+b1; BN; project; pool
    hipMemsetAsync(agg, 0, (size_t)N * 16 * 4, stream);
    edge_scatter_kernel<<<((size_t)E * 4 + 255) / 256, 256, 0, stream>>>(ei, h0, agg, E);
    gin_pre_kernel<16><<<NB, 256, 0, stream>>>(h0, agg, W1, b1, pre, partE, N);
    bn_finalize_kernel<<<1, 32, 0, stream>>>(partE, NB, 16, invN, g1, be1, ab1);
    layer_post_kernel<16, false><<<NB, 256, 0, stream>>>(
        pre, ab1, Wl1, bl1, bat, h1, out_zsum, pool3 + 128, N);

    // layer 2
    hipMemsetAsync(agg, 0, (size_t)N * 16 * 4, stream);
    edge_scatter_kernel<<<((size_t)E * 4 + 255) / 256, 256, 0, stream>>>(ei, h1, agg, E);
    gin_pre_kernel<8><<<NB, 256, 0, stream>>>(h1, agg, W2, b2, pre, partE, N);
    bn_finalize_kernel<<<1, 32, 0, stream>>>(partE, NB, 8, invN, g2, be2, ab2);
    layer_post_kernel<8, false><<<NB, 256, 0, stream>>>(
        pre, ab2, Wl2, bl2, bat, out_h, out_zsum, pool3 + 256, N);

    // combine pooled maxes -> out[16,8]
    pool_combine_kernel<<<1, 128, 0, stream>>>(pool3, out_pool);
}
